// EnvironVectorField_55929064129218
// MI455X (gfx1250) — matrix operand order repaired and checked by reading, not compiled, not run on hardware
//
#include <hip/hip_runtime.h>

// Fused grouped-MLP (8 envs, 64->1024->1024->1024->64, swish) for gfx1250.
// v3: like v2 (ping-pong LDS activation buffers, 2 x 64x1032 f16 = 264KB; one barrier
//     per layer; 64-VGPR accumulator chunks; rcp-based swish) plus '#pragma unroll 2'
//     on the K loops so two k-steps of B(global/L2) + A(LDS) loads are in flight per
//     wave, overlapping WMMA with L2 latency without recreating Round-1's spills.
// Weights pre-converted to f16 + transposed (N-major) in d_ws (~36MB, L2-resident).

typedef __attribute__((ext_vector_type(16))) _Float16 v16h;
typedef __attribute__((ext_vector_type(8)))  float    v8f;

#define N_ENVS 8
#define SDIM   64
#define HID    1024
#define BATCHP 16384          // rows per env
#define MT     64             // batch rows per block tile
#define RS     (HID + 8)      // LDS row stride in halves (1032: 16B multiple, no 64-bank conflicts)
#define NWAVES 16
#define BLOCK  (NWAVES * 32)

// ---- weight convert + transpose: W (E, IN, OUT) f32 -> Wt (E, OUT, IN) f16 ----
__global__ void cvt_transpose_f16(const float* __restrict__ W, _Float16* __restrict__ Wt,
                                  int IN, int OUT, int total) {
  int idx = blockIdx.x * blockDim.x + threadIdx.x;
  if (idx >= total) return;
  int per = IN * OUT;
  int e = idx / per;
  int r = idx - e * per;
  int i = r / OUT;
  int o = r - i * OUT;
  Wt[((size_t)e * OUT + o) * IN + i] = (_Float16)W[idx];
}

// Assemble a 16-half fragment from two 16B chunks.
__device__ __forceinline__ v16h ld_frag(const _Float16* p0, const _Float16* p1) {
  union { v16h v; float4 f[2]; } x;
  x.f[0] = *(const float4*)p0;
  x.f[1] = *(const float4*)p1;
  return x.v;
}

__device__ __forceinline__ float fast_swish(float v) {
  // v * sigmoid(v); approximate rcp is plenty for an activation function.
  return v * __builtin_amdgcn_rcpf(1.0f + __expf(-v));
}

// One hidden layer, ping-pong: src(64xK f16, LDS) x Wt(N-major f16, L2) -> swish -> dst (LDS).
// Each wave computes NPASS sequential 64x(NT*16) chunks; acc = 4*NT*8 VGPRs per pass.
// Writes go to dst while reads come from src, so no intra-layer barrier is needed.
template <int NT, int NPASS>
__device__ __forceinline__ void layer_pp(const _Float16* __restrict__ src,
                                         _Float16* __restrict__ dst,
                                         const _Float16* __restrict__ Wt,
                                         const float* __restrict__ bias,
                                         int K, int lane, int wave) {
  const int l15  = lane & 15;
  const int hsel = lane >> 4;            // 0: lanes 0-15, 1: lanes 16-31
  // A-frag base: lanes 0-15 row M=l15 hold K {k..k+7, k+16..k+23}; lanes 16-31 shifted by 8.
  const _Float16* abase = src + (size_t)l15 * RS + hsel * 8;

  for (int p = 0; p < NPASS; ++p) {
    const int n_base = (wave * NPASS + p) * (NT * 16);

    v8f acc[4][NT];
#pragma unroll
    for (int mt = 0; mt < 4; ++mt)
#pragma unroll
      for (int nt = 0; nt < NT; ++nt)
#pragma unroll
        for (int i = 0; i < 8; ++i) acc[mt][nt][i] = 0.0f;

#pragma unroll 2
    for (int k = 0; k < K; k += 32) {
      v16h afr[4];
#pragma unroll
      for (int mt = 0; mt < 4; ++mt) {
        const _Float16* ap = abase + (size_t)(mt * 16) * RS + k;
        afr[mt] = ld_frag(ap, ap + 16);
      }
      v16h bfr[NT];
#pragma unroll
      for (int nt = 0; nt < NT; ++nt) {
        // B layout: lane = column N; lanes 0-15 hold K {k..k+7, k+16..k+23}; lanes 16-31 shifted by 8.
        const _Float16* bp = Wt + (size_t)(n_base + nt * 16 + l15) * K + k + hsel * 8;
        bfr[nt] = ld_frag(bp, bp + 16);
      }
#pragma unroll
      for (int mt = 0; mt < 4; ++mt)
#pragma unroll
        for (int nt = 0; nt < NT; ++nt)
          acc[mt][nt] = __builtin_amdgcn_wmma_f32_16x16x32_f16(
              false, afr[mt], false, bfr[nt], (short)0, acc[mt][nt], false, false);
    }

#pragma unroll
    for (int nt = 0; nt < NT; ++nt) {
      const int col = n_base + nt * 16 + l15;
      const float bv = bias[col];
#pragma unroll
      for (int mt = 0; mt < 4; ++mt)
#pragma unroll
        for (int i = 0; i < 8; ++i) {
          // C/D layout: VGPR i -> M = i + 8*hsel within the 16x16 tile, N = l15.
          float v = fast_swish(acc[mt][nt][i] + bv);
          dst[(size_t)(mt * 16 + hsel * 8 + i) * RS + col] = (_Float16)v;
        }
    }
  }
}

__global__ void __launch_bounds__(BLOCK)
mlp_fused_kernel(const float* __restrict__ u,
                 const float* __restrict__ b1, const float* __restrict__ b2,
                 const float* __restrict__ b3, const float* __restrict__ b4,
                 const _Float16* __restrict__ w1t, const _Float16* __restrict__ w2t,
                 const _Float16* __restrict__ w3t, const _Float16* __restrict__ w4t,
                 float* __restrict__ out) {
  extern __shared__ _Float16 lds[];      // 2 x MT x RS halves = 264KB
  _Float16* buf0 = lds;
  _Float16* buf1 = lds + (size_t)MT * RS;

  const int env = blockIdx.y;
  const int b0  = blockIdx.x * MT;       // batch row base within env
  const int tid  = threadIdx.x;
  const int lane = tid & 31;
  const int wave = tid >> 5;

  // Stage input tile (env rows interleaved: global row = b*8 + env), float4-vectorized.
  for (int idx = tid; idx < MT * (SDIM / 4); idx += BLOCK) {
    int m  = idx >> 4;                   // 16 float4 per row
    int c4 = idx & 15;
    const float4 v4 = *(const float4*)(u + ((size_t)(b0 + m) * N_ENVS + env) * SDIM + c4 * 4);
    union { _Float16 h[4]; float2 f2; } pk;
    pk.h[0] = (_Float16)v4.x; pk.h[1] = (_Float16)v4.y;
    pk.h[2] = (_Float16)v4.z; pk.h[3] = (_Float16)v4.w;
    *(float2*)(buf0 + (size_t)m * RS + c4 * 4) = pk.f2;
  }
  __syncthreads();

  layer_pp<2, 2>(buf0, buf1, w1t + (size_t)env * HID * SDIM, b1 + (size_t)env * HID, SDIM, lane, wave);
  __syncthreads();
  layer_pp<2, 2>(buf1, buf0, w2t + (size_t)env * HID * HID,  b2 + (size_t)env * HID, HID,  lane, wave);
  __syncthreads();
  layer_pp<2, 2>(buf0, buf1, w3t + (size_t)env * HID * HID,  b3 + (size_t)env * HID, HID,  lane, wave);
  __syncthreads();

  // Layer 4: 64x1024 @ 1024x64 + b4, no activation, FACTOR==1.0. Waves 0..3 cover N=64.
  if (wave < 4) {
    const int l15  = lane & 15;
    const int hsel = lane >> 4;
    const int n_base = wave * 16;
    const _Float16* Wt = w4t + (size_t)env * SDIM * HID;
    const _Float16* abase = buf1 + (size_t)l15 * RS + hsel * 8;

    v8f acc[4];
#pragma unroll
    for (int mt = 0; mt < 4; ++mt)
#pragma unroll
      for (int i = 0; i < 8; ++i) acc[mt][i] = 0.0f;

#pragma unroll 2
    for (int k = 0; k < HID; k += 32) {
      v16h afr[4];
#pragma unroll
      for (int mt = 0; mt < 4; ++mt) {
        const _Float16* ap = abase + (size_t)(mt * 16) * RS + k;
        afr[mt] = ld_frag(ap, ap + 16);
      }
      const _Float16* bp = Wt + (size_t)(n_base + l15) * HID + k + hsel * 8;
      v16h bfr = ld_frag(bp, bp + 16);
#pragma unroll
      for (int mt = 0; mt < 4; ++mt)
        acc[mt] = __builtin_amdgcn_wmma_f32_16x16x32_f16(
            false, afr[mt], false, bfr, (short)0, acc[mt], false, false);
    }

    const int col = n_base + l15;
    const float bv = b4[(size_t)env * SDIM + col];
#pragma unroll
    for (int mt = 0; mt < 4; ++mt)
#pragma unroll
      for (int i = 0; i < 8; ++i) {
        int m = mt * 16 + hsel * 8 + i;
        out[((size_t)(b0 + m) * N_ENVS + env) * SDIM + col] = acc[mt][i] + bv;
      }
  }
}

extern "C" void kernel_launch(void* const* d_in, const int* in_sizes, int n_in,
                              void* d_out, int out_size, void* d_ws, size_t ws_size,
                              hipStream_t stream) {
  (void)in_sizes; (void)n_in; (void)out_size; (void)ws_size;
  // dict order: t, u, W1, b1, W2, b2, W3, b3, W4, b4
  const float* u  = (const float*)d_in[1];
  const float* W1 = (const float*)d_in[2];
  const float* b1 = (const float*)d_in[3];
  const float* W2 = (const float*)d_in[4];
  const float* b2 = (const float*)d_in[5];
  const float* W3 = (const float*)d_in[6];
  const float* b3 = (const float*)d_in[7];
  const float* W4 = (const float*)d_in[8];
  const float* b4 = (const float*)d_in[9];
  float* out = (float*)d_out;

  // Workspace layout (f16 transposed weights, ~35.7MB total):
  _Float16* w1t = (_Float16*)d_ws;
  _Float16* w2t = w1t + (size_t)N_ENVS * SDIM * HID;
  _Float16* w3t = w2t + (size_t)N_ENVS * HID * HID;
  _Float16* w4t = w3t + (size_t)N_ENVS * HID * HID;

  const int t1 = N_ENVS * SDIM * HID;   // 524288
  const int t2 = N_ENVS * HID * HID;    // 8388608
  cvt_transpose_f16<<<(t1 + 255) / 256, 256, 0, stream>>>(W1, w1t, SDIM, HID, t1);
  cvt_transpose_f16<<<(t2 + 255) / 256, 256, 0, stream>>>(W2, w2t, HID, HID, t2);
  cvt_transpose_f16<<<(t2 + 255) / 256, 256, 0, stream>>>(W3, w3t, HID, HID, t2);
  cvt_transpose_f16<<<(t1 + 255) / 256, 256, 0, stream>>>(W4, w4t, HID, SDIM, t1);

  size_t shmem = 2 * (size_t)MT * RS * sizeof(_Float16);  // 264192 bytes (<320KB/WGP)
  (void)hipFuncSetAttribute((const void*)mlp_fused_kernel,
                            hipFuncAttributeMaxDynamicSharedMemorySize, (int)shmem);
  dim3 grid(BATCHP / MT, N_ENVS);
  mlp_fused_kernel<<<grid, BLOCK, shmem, stream>>>(u, b1, b2, b3, b4,
                                                   w1t, w2t, w3t, w4t, out);
}
